// MultiHeadSelfAttention_19516331393206
// MI455X (gfx1250) — matrix operand order repaired and checked by reading, not compiled, not run on hardware
//
#include <hip/hip_runtime.h>
#include <hip/hip_bf16.h>
#include <stdint.h>

typedef __attribute__((ext_vector_type(16))) _Float16 v16h;
typedef __attribute__((ext_vector_type(8)))  _Float16 v8h;
typedef __attribute__((ext_vector_type(4)))  _Float16 v4h;
typedef __attribute__((ext_vector_type(8)))  float    v8f;
typedef __attribute__((ext_vector_type(4)))  float    v4f;

#define B_  4
#define N_  2048
#define C_  1024
#define H_  16
#define D_  64
#define MTOT (B_*N_)   // 8192

union Frag16 { v16h v; v8h h[2]; };

// ---- WMMA wrapper: D = A(16x32 f16) x B(32x16 f16) + C(16x16 f32) ----
__device__ inline v8f wmma_f16(v16h a, v16h b, v8f c) {
  return __builtin_amdgcn_wmma_f32_16x16x32_f16(
      /*neg_a=*/false, a, /*neg_b=*/false, b,
      /*c_mod=*/(short)0, c, /*reuse_a=*/false, /*reuse_b=*/false);
}

// A-fragment (16x32, 16-bit) per ISA 7.12.2:
// lane&15 = row; lanes<16 hold K {half*8..+7} in v0..3 and {16+half*8..} in v4..7
__device__ inline v16h load_frag_a(const _Float16* __restrict__ base, int ld, int lane) {
  int r = lane & 15, hf = lane >> 4;
  const _Float16* p = base + (size_t)r * ld + hf * 8;
  Frag16 f;
  f.h[0] = *(const v8h*)(p);        // K = hf*8 .. hf*8+7
  f.h[1] = *(const v8h*)(p + 16);   // K = 16+hf*8 ..
  return f.v;
}

// B-fragment (32x16, 16-bit), loaded from B^T stored row-major (i.e. N x K):
// lane&15 = column n of B; lanes<16 hold K {half*8..+7} then {16+half*8..+7} (same K order as A)
__device__ inline v16h load_frag_b(const _Float16* __restrict__ baseT, int ld, int lane) {
  int c = lane & 15, hf = lane >> 4;
  const _Float16* p = baseT + (size_t)c * ld + hf * 8;
  Frag16 f;
  f.h[0] = *(const v8h*)(p);
  f.h[1] = *(const v8h*)(p + 16);
  return f.v;
}

// CDNA5 async DMA: global -> LDS, 16B per lane, tracked by ASYNCcnt.
// VDST = 32-bit LDS byte address (low 32 bits of generic shared pointer, per
// ISA flat->LDS mapping), VADDR = 64-bit global address.
__device__ inline void async_copy_b128(void* lds_dst, const void* gsrc) {
  uint32_t lds_addr = (uint32_t)(uintptr_t)lds_dst;
  asm volatile("global_load_async_to_lds_b128 %0, %1, off"
               :: "v"(lds_addr), "v"(gsrc) : "memory");
}
// async loads retire in order -> threshold waits give true double buffering
__device__ inline void wait_async_le4() {
  asm volatile("s_wait_asynccnt 0x4" ::: "memory");
}
__device__ inline void wait_async_le0() {
  asm volatile("s_wait_asynccnt 0x0" ::: "memory");
}

// ---------------- f32 -> f16 conversion (vectorized x4) ----------------
__global__ __launch_bounds__(256) void cvt_f32_f16(const float* __restrict__ in,
                                                   _Float16* __restrict__ out, int n) {
  int i = (blockIdx.x * blockDim.x + threadIdx.x) * 4;
  if (i + 3 < n) {
    v4f a = *(const v4f*)(in + i);
    v4h o;
    o[0] = (_Float16)a[0]; o[1] = (_Float16)a[1];
    o[2] = (_Float16)a[2]; o[3] = (_Float16)a[3];
    *(v4h*)(out + i) = o;
  }
}

// ---------------- QKV GEMM: [8192x1024] x Wqkv^T -> scatter q,k,vT (f16) ----------------
// block = 256 threads (8 waves: 4 M x 2 N), block tile 128x128, wave tile 32x64, K step 32
__global__ __launch_bounds__(256) void qkv_gemm(const _Float16* __restrict__ xh,
                                                const _Float16* __restrict__ wh,   // [3072x1024] row-major (= B^T)
                                                const float* __restrict__ bias,    // [3072]
                                                _Float16* __restrict__ q,
                                                _Float16* __restrict__ k,
                                                _Float16* __restrict__ vT) {
  int lane = threadIdx.x & 31, w = threadIdx.x >> 5;
  int wm = w & 3, wn = w >> 2;
  int row0 = blockIdx.x * 128 + wm * 32;
  int col0 = blockIdx.y * 128 + wn * 64;
  v8f acc[2][4] = {};
  for (int kt = 0; kt < C_; kt += 32) {
    __builtin_prefetch(xh + (size_t)row0 * C_ + kt + 64, 0, 0);
    v16h a0 = load_frag_a(xh + (size_t)row0 * C_ + kt, C_, lane);
    v16h a1 = load_frag_a(xh + (size_t)(row0 + 16) * C_ + kt, C_, lane);
#pragma unroll
    for (int ni = 0; ni < 4; ++ni) {
      v16h bf = load_frag_b(wh + (size_t)(col0 + ni * 16) * C_ + kt, C_, lane);
      acc[0][ni] = wmma_f16(a0, bf, acc[0][ni]);
      acc[1][ni] = wmma_f16(a1, bf, acc[1][ni]);
    }
  }
  int hf = lane >> 4, nn = lane & 15;
#pragma unroll
  for (int mi = 0; mi < 2; ++mi)
#pragma unroll
    for (int ni = 0; ni < 4; ++ni)
#pragma unroll
      for (int e = 0; e < 8; ++e) {
        int gr = row0 + mi * 16 + e + 8 * hf;   // C-layout: m = e + 8*(lane>=16)
        int gc = col0 + ni * 16 + nn;           // n = lane&15
        float val = acc[mi][ni][e] + bias[gc];
        int slot = gc >> 10, ci = gc & 1023;
        int bb = gr >> 11, ns = gr & 2047;
        int hh = ci >> 6, dd = ci & 63;
        size_t bhi = (size_t)bb * H_ + hh;
        if (slot == 0)      q[(bhi * N_ + ns) * D_ + dd] = (_Float16)(val * 0.125f); // fold 1/sqrt(D)
        else if (slot == 1) k[(bhi * N_ + ns) * D_ + dd] = (_Float16)val;
        else                vT[(bhi * D_ + dd) * N_ + ns] = (_Float16)val;           // v transposed
      }
}

// ---------------- Flash attention: per (b,h), 128 queries/block, 64-key blocks ----------------
// Double-buffered async-DMA staging: next K/V^T block streams into LDS while
// WMMA/softmax runs on the current one (ASYNCcnt threshold wait, in-order retire).
__global__ __launch_bounds__(256) void flash_attn(const _Float16* __restrict__ q,
                                                  const _Float16* __restrict__ k,
                                                  const _Float16* __restrict__ vT,
                                                  _Float16* __restrict__ attnout) {
  __shared__ _Float16 ldsK[2][64 * 64];    // key block   [key][d]
  __shared__ _Float16 ldsV[2][64 * 64];    // vT block    [d][key]
  __shared__ _Float16 ldsP[8][16 * 64];    // per-wave P tile bounce (C-layout -> A-layout)

  int lane = threadIdx.x & 31, w = threadIdx.x >> 5;
  int hf = lane >> 4, nn = lane & 15;
  int bh = blockIdx.x;
  int q0 = blockIdx.y * 128 + w * 16;
  const _Float16* qbase = q  + (size_t)bh * N_ * D_;
  const _Float16* kbase = k  + (size_t)bh * N_ * D_;
  const _Float16* vbase = vT + (size_t)bh * D_ * N_;

  v16h aq[2];
  aq[0] = load_frag_a(qbase + (size_t)q0 * D_ + 0,  D_, lane);
  aq[1] = load_frag_a(qbase + (size_t)q0 * D_ + 32, D_, lane);

  v8f o[4] = {};
  float mr[8], lr[8];
#pragma unroll
  for (int i = 0; i < 8; ++i) { mr[i] = -1.0e30f; lr[i] = 0.0f; }

  // stage one (b,h) K/V^T block: 4 async b128 ops per wave (2 iters x {K,V})
  auto stage_kv = [&](int buf, int kb) {
#pragma unroll
    for (int c = threadIdx.x; c < 512; c += 256) {
      int rr = c >> 3, cc = (c & 7) * 8;
      async_copy_b128(&ldsK[buf][rr * 64 + cc], kbase + (size_t)(kb + rr) * D_ + cc);
      async_copy_b128(&ldsV[buf][rr * 64 + cc], vbase + (size_t)rr * N_ + kb + cc);
    }
  };

  stage_kv(0, 0);  // prologue

  for (int kb = 0; kb < N_; kb += 64) {
    int cur = (kb >> 6) & 1;
    if (kb + 64 < N_) {
      // kick off next block's DMA (buffer cur^1 was last read two barriers ago),
      // then wait only for the CURRENT block's 4 ops (in-order completion).
      stage_kv(cur ^ 1, kb + 64);
      wait_async_le4();
    } else {
      wait_async_le0();
    }
    __syncthreads();

    // S = q * k^T  (16 queries x 64 keys), f32 accum
    v8f s[4] = {};
#pragma unroll
    for (int j = 0; j < 2; ++j)
#pragma unroll
      for (int t = 0; t < 4; ++t) {
        v16h bf = load_frag_b(&ldsK[cur][t * 16 * 64 + j * 32], 64, lane);
        s[t] = wmma_f16(aq[j], bf, s[t]);
      }

    // online softmax: row max across 4 tiles, then across 16 lanes of this half
    float mnew[8];
#pragma unroll
    for (int i = 0; i < 8; ++i) {
      mnew[i] = mr[i];
#pragma unroll
      for (int t = 0; t < 4; ++t) mnew[i] = fmaxf(mnew[i], s[t][i]);
#pragma unroll
      for (int off = 1; off < 16; off <<= 1)
        mnew[i] = fmaxf(mnew[i], __shfl_xor(mnew[i], off, 32));
    }
    float al[8], rs[8];
    v8f pf[4];
#pragma unroll
    for (int i = 0; i < 8; ++i) al[i] = __expf(mr[i] - mnew[i]);
#pragma unroll
    for (int t = 0; t < 4; ++t)
#pragma unroll
      for (int i = 0; i < 8; ++i) pf[t][i] = __expf(s[t][i] - mnew[i]);
#pragma unroll
    for (int i = 0; i < 8; ++i) {
      rs[i] = pf[0][i] + pf[1][i] + pf[2][i] + pf[3][i];
#pragma unroll
      for (int off = 1; off < 16; off <<= 1)
        rs[i] += __shfl_xor(rs[i], off, 32);
      lr[i] = lr[i] * al[i] + rs[i];
      mr[i] = mnew[i];
    }
#pragma unroll
    for (int t = 0; t < 4; ++t)
#pragma unroll
      for (int i = 0; i < 8; ++i) o[t][i] *= al[i];

    // bounce P through per-wave LDS: C-layout store -> A-layout load (same-wave DS is in-order)
    _Float16* pbuf = &ldsP[w][0];
#pragma unroll
    for (int t = 0; t < 4; ++t)
#pragma unroll
      for (int i = 0; i < 8; ++i)
        pbuf[(i + 8 * hf) * 64 + t * 16 + nn] = (_Float16)pf[t][i];
    asm volatile("" ::: "memory");  // keep DS store->load order at IR level
    v16h pa0 = load_frag_a(pbuf + 0,  64, lane);
    v16h pa1 = load_frag_a(pbuf + 32, 64, lane);

    // O += P * V   (B-fragments come from vT tile: contiguous in key dim)
#pragma unroll
    for (int t = 0; t < 4; ++t) {
      v16h bv0 = load_frag_b(&ldsV[cur][t * 16 * 64 + 0],  64, lane);
      v16h bv1 = load_frag_b(&ldsV[cur][t * 16 * 64 + 32], 64, lane);
      o[t] = wmma_f16(pa0, bv0, o[t]);
      o[t] = wmma_f16(pa1, bv1, o[t]);
    }
    __syncthreads();  // protects buffer reuse two iterations ahead
  }

  // epilogue: normalize and store to [B, N, H*D] as f16
  int bb = bh >> 4, hh = bh & 15;
#pragma unroll
  for (int t = 0; t < 4; ++t)
#pragma unroll
    for (int i = 0; i < 8; ++i) {
      int m = i + 8 * hf;
      float val = o[t][i] / lr[i];
      attnout[(size_t)(bb * N_ + q0 + m) * C_ + hh * D_ + t * 16 + nn] = (_Float16)val;
    }
}

// ---------------- Output projection: attn[8192x1024] x Wproj^T + b -> f32 out ----------------
__global__ __launch_bounds__(256) void proj_gemm(const _Float16* __restrict__ ah,
                                                 const _Float16* __restrict__ wh,  // [1024x1024] row-major
                                                 const float* __restrict__ bias,
                                                 float* __restrict__ out) {
  int lane = threadIdx.x & 31, w = threadIdx.x >> 5;
  int wm = w & 3, wn = w >> 2;
  int row0 = blockIdx.x * 128 + wm * 32;
  int col0 = blockIdx.y * 128 + wn * 64;
  v8f acc[2][4] = {};
  for (int kt = 0; kt < C_; kt += 32) {
    __builtin_prefetch(ah + (size_t)row0 * C_ + kt + 64, 0, 0);
    v16h a0 = load_frag_a(ah + (size_t)row0 * C_ + kt, C_, lane);
    v16h a1 = load_frag_a(ah + (size_t)(row0 + 16) * C_ + kt, C_, lane);
#pragma unroll
    for (int ni = 0; ni < 4; ++ni) {
      v16h bf = load_frag_b(wh + (size_t)(col0 + ni * 16) * C_ + kt, C_, lane);
      acc[0][ni] = wmma_f16(a0, bf, acc[0][ni]);
      acc[1][ni] = wmma_f16(a1, bf, acc[1][ni]);
    }
  }
  int hf = lane >> 4, nn = lane & 15;
#pragma unroll
  for (int mi = 0; mi < 2; ++mi)
#pragma unroll
    for (int ni = 0; ni < 4; ++ni)
#pragma unroll
      for (int e = 0; e < 8; ++e) {
        int gr = row0 + mi * 16 + e + 8 * hf;
        int gc = col0 + ni * 16 + nn;
        out[(size_t)gr * C_ + gc] = acc[mi][ni][e] + bias[gc];
      }
}

extern "C" void kernel_launch(void* const* d_in, const int* in_sizes, int n_in,
                              void* d_out, int out_size, void* d_ws, size_t ws_size,
                              hipStream_t stream) {
  (void)in_sizes; (void)n_in; (void)out_size; (void)ws_size;
  const float* x      = (const float*)d_in[0];   // [4,2048,1024]
  const float* w_qkv  = (const float*)d_in[1];   // [3072,1024]
  const float* b_qkv  = (const float*)d_in[2];   // [3072]
  const float* w_proj = (const float*)d_in[3];   // [1024,1024]
  const float* b_proj = (const float*)d_in[4];   // [1024]
  float* out = (float*)d_out;

  // workspace carve-up (all f16, sizes in elements)
  const size_t SZ_XH   = (size_t)MTOT * C_;       // 8192*1024
  const size_t SZ_WQKV = (size_t)3 * C_ * C_;     // 3072*1024
  const size_t SZ_WPRJ = (size_t)C_ * C_;
  const size_t SZ_HEAD = (size_t)B_ * H_ * N_ * D_;
  char* p = (char*)d_ws;
  _Float16* xh    = (_Float16*)p; p += SZ_XH   * 2;
  _Float16* wqkvh = (_Float16*)p; p += SZ_WQKV * 2;
  _Float16* wprjh = (_Float16*)p; p += SZ_WPRJ * 2;
  _Float16* qd    = (_Float16*)p; p += SZ_HEAD * 2;
  _Float16* kd    = (_Float16*)p; p += SZ_HEAD * 2;
  _Float16* vTd   = (_Float16*)p; p += SZ_HEAD * 2;
  _Float16* attn  = (_Float16*)p; p += SZ_XH   * 2;

  // 1) convert inputs/weights to f16
  cvt_f32_f16<<<(int)(SZ_XH   / 4 / 256), 256, 0, stream>>>(x,      xh,    (int)SZ_XH);
  cvt_f32_f16<<<(int)(SZ_WQKV / 4 / 256), 256, 0, stream>>>(w_qkv,  wqkvh, (int)SZ_WQKV);
  cvt_f32_f16<<<(int)(SZ_WPRJ / 4 / 256), 256, 0, stream>>>(w_proj, wprjh, (int)SZ_WPRJ);

  // 2) QKV projection + head scatter (q scaled by 1/sqrt(D), v transposed)
  qkv_gemm<<<dim3(MTOT / 128, (3 * C_) / 128), 256, 0, stream>>>(xh, wqkvh, b_qkv, qd, kd, vTd);

  // 3) flash attention per (b,h)
  flash_attn<<<dim3(B_ * H_, N_ / 128), 256, 0, stream>>>(qd, kd, vTd, attn);

  // 4) output projection (f32 out + bias)
  proj_gemm<<<dim3(MTOT / 128, C_ / 128), 256, 0, stream>>>(attn, wprjh, b_proj, out);
}
